// GraphVariationalAutoencoder_3504693314185
// MI455X (gfx1250) — matrix operand order repaired and checked by reading, not compiled, not run on hardware
//
#include <hip/hip_runtime.h>
#include <math.h>

// ---------------------------------------------------------------------------
// Types for CDNA5 WMMA (wave32)
// ---------------------------------------------------------------------------
typedef __attribute__((ext_vector_type(16))) __bf16          bf16x16;
typedef __attribute__((ext_vector_type(8)))  float           floatx8;
typedef __attribute__((ext_vector_type(2)))  unsigned int    uint2v;

// Round-half-up f32 -> bf16 (high 16 bits). One v_add; bias negligible for
// 8-bit-mantissa data, accumulation stays f32.
__device__ __forceinline__ unsigned short f2bf_h(float x) {
    return (unsigned short)((__float_as_uint(x) + 0x8000u) >> 16);
}
// Pack two f32 -> packed 2xbf16 in one v_perm_b32 (selects high halves).
__device__ __forceinline__ unsigned pk_bf16(float a, float b) {
    const unsigned ua = __float_as_uint(a) + 0x8000u;
    const unsigned ub = __float_as_uint(b) + 0x8000u;
    return __builtin_amdgcn_perm(ub, ua, 0x07060302u);
}

// ---------------------------------------------------------------------------
// Generic gathered GEMM:  C = act( A_eff @ Bm + bias )
//   A_eff[i,j] = A[(RG?ridx[i]:i)*lda + (CG?cidx[j]:j)]   (f32, L2-resident)
//   Bm: [K, Ncols] f32 row-major (ldb), C: [M, Ncols] f32 (ldc)
//   Per-batch (blockIdx.z): A += z*strideA, ridx += z*M, cidx += z*K, etc.
// Tiling: 128x64 per 256-thread block (8 waves), K-step 32,
//         bf16 WMMA (v_wmma_f32_16x16x32_bf16) with f32 accumulation.
// Software-pipelined: global loads of tile t+1 overlap WMMAs of tile t.
// Row offsets pre-multiplied + register-hoisted; conversions via v_perm_b32.
// ACT: 0=none, 1=relu, 2=softplus.  RG/CG: row/col gather of A via indices.
// Requires: M % 128 == 0, K % 32 == 0, Ncols % 4 == 0, lda/ldb/ldc % 4 == 0.
// ---------------------------------------------------------------------------
template <int ACT, bool RG, bool CG>
__global__ __launch_bounds__(256)
void gemm_bf16_wmma(const float* __restrict__ A, long strideA, int lda,
                    const int* __restrict__ ridx, const int* __restrict__ cidx,
                    const float* __restrict__ Bm, long strideB, int ldb,
                    const float* __restrict__ bias,
                    float* __restrict__ C, long strideC, int ldc,
                    int M, int K, int Ncols)
{
    const int z = blockIdx.z;
    A  += (long)z * strideA;
    Bm += (long)z * strideB;
    C  += (long)z * strideC;
    const int* cix = CG ? (cidx + (long)z * K) : nullptr;

    const int row0 = blockIdx.x * 128;
    const int n0   = blockIdx.y * 64;
    const int tid  = threadIdx.x;
    const int w    = tid >> 5;    // wave 0..7 -> 16-row strip
    const int lane = tid & 31;

    // LDS tiles stored directly in WMMA fragment order:
    //   sA[(mb*32 + lane)*16 + e], sB[(nb*32 + lane)*16 + e]
    __shared__ __align__(32) unsigned short sA[8 * 32 * 16]; // 128x32 bf16
    __shared__ __align__(32) unsigned short sB[4 * 32 * 16]; // 32x64  bf16
    __shared__ int sRO[128];                                 // row offsets (r*lda)

    if (tid < 128) {
        const int r = RG ? ridx[(long)z * M + row0 + tid] : (row0 + tid);
        sRO[tid] = r * lda;
    }
    __syncthreads();

    // ---- hoist per-thread row offsets into registers (once) ----------------
    unsigned rowOffQ[4];   // quad-load path (!CG): rows (i*256+tid)>>3
    unsigned rowOffC[16];  // column-gather path (CG): rows i*8 + (tid>>5)
    if (CG) {
#pragma unroll
        for (int i = 0; i < 16; ++i) rowOffC[i] = (unsigned)sRO[i * 8 + (tid >> 5)];
    } else {
#pragma unroll
        for (int i = 0; i < 4; ++i) rowOffQ[i] = (unsigned)sRO[(i * 256 + tid) >> 3];
    }

    // ---- per-thread B-tile constants ---------------------------------------
    unsigned bOff[2]; bool bIn[2];
#pragma unroll
    for (int i = 0; i < 2; ++i) {
        const int q  = i * 256 + tid;         // 512 quads
        const int kk = q >> 4;
        const int nq = (q & 15) * 4;
        bIn[i]  = (n0 + nq) < Ncols;
        bOff[i] = (unsigned)(kk * ldb + n0 + nq);
    }

    // per-thread constants for the column-gather A loader
    const int kkA = tid & 31;                                 // fixed k in tile
    const int chA = (kkA >> 3) & 1;                           // lane half
    const int ceA = (kkA & 7) | ((kkA >> 4) << 3);            // fragment elem

    floatx8 acc[4];
#pragma unroll
    for (int nb = 0; nb < 4; ++nb)
#pragma unroll
        for (int v = 0; v < 8; ++v) acc[nb][v] = 0.0f;

    // ---- register staging for the software pipeline ------------------------
    float  avalC[16];   // CG path
    float4 aq[4];       // !CG path
    float4 bq[2];

    auto load_tiles = [&](int kt) {
        if (CG) {
            const unsigned ca = (unsigned)cix[kt + kkA];      // one index read
#pragma unroll
            for (int i = 0; i < 16; ++i)
                avalC[i] = A[rowOffC[i] + ca];
        } else {
#pragma unroll
            for (int i = 0; i < 4; ++i) {
                const int q  = i * 256 + tid;
                const int kq = (q & 7) * 4;
                aq[i] = *(const float4*)(A + (rowOffQ[i] + (unsigned)(kt + kq)));
            }
        }
        const unsigned kb = (unsigned)(kt * ldb);
#pragma unroll
        for (int i = 0; i < 2; ++i) {
            if (bIn[i]) bq[i] = *(const float4*)(Bm + (kb + bOff[i]));
            else { bq[i].x = 0.f; bq[i].y = 0.f; bq[i].z = 0.f; bq[i].w = 0.f; }
        }
    };

    auto store_tiles = [&]() {
        if (CG) {
#pragma unroll
            for (int i = 0; i < 16; ++i) {
                const int mm = i * 8 + (tid >> 5);
                sA[((mm >> 4) * 32 + chA * 16 + (mm & 15)) * 16 + ceA] = f2bf_h(avalC[i]);
            }
        } else {
#pragma unroll
            for (int i = 0; i < 4; ++i) {
                const int q  = i * 256 + tid;
                const int mm = q >> 3;
                const int kq = (q & 7) * 4;
                uint2v pk;
                pk.x = pk_bf16(aq[i].x, aq[i].y);
                pk.y = pk_bf16(aq[i].z, aq[i].w);
                const int half = (kq >> 3) & 1;
                const int e0   = (kq & 7) | ((kq >> 4) << 3);
                *(uint2v*)(sA + ((mm >> 4) * 32 + half * 16 + (mm & 15)) * 16 + e0) = pk;
            }
        }
#pragma unroll
        for (int i = 0; i < 2; ++i) {
            const int q    = i * 256 + tid;
            const int kk   = q >> 4;
            const int nq   = (q & 15) * 4;
            const int base = ((nq >> 4) * 32 + ((kk >> 3) & 1) * 16 + (nq & 15)) * 16 + ((kk & 7) | ((kk >> 4) << 3));
            sB[base]      = f2bf_h(bq[i].x);
            sB[base + 16] = f2bf_h(bq[i].y);
            sB[base + 32] = f2bf_h(bq[i].z);
            sB[base + 48] = f2bf_h(bq[i].w);
        }
    };

    // ---- pipelined main loop ----------------------------------------------
    load_tiles(0);
    for (int k0 = 0; k0 < K; k0 += 32) {
        store_tiles();
        __syncthreads();
        if (k0 + 32 < K) load_tiles(k0 + 32);   // overlaps with WMMAs below

        const bf16x16 afrag = *(const bf16x16*)(sA + (w * 32 + lane) * 16);
#pragma unroll
        for (int nb = 0; nb < 4; ++nb) {
            const bf16x16 bfrag = *(const bf16x16*)(sB + (nb * 32 + lane) * 16);
            acc[nb] = __builtin_amdgcn_wmma_f32_16x16x32_bf16(
                false, afrag, false, bfrag, (short)0, acc[nb], false, false);
        }
        __syncthreads();
    }

    // ---- epilogue: C/D layout -> M = w*16 + (lane/16)*8 + v, N = nb*16 + lane%16
    const int mrow  = row0 + w * 16 + (lane >> 4) * 8;
    const int ncol0 = n0 + (lane & 15);
#pragma unroll
    for (int nb = 0; nb < 4; ++nb) {
        const int gn = ncol0 + nb * 16;
        if (gn < Ncols) {
            const float bv = bias ? bias[gn] : 0.0f;
#pragma unroll
            for (int v = 0; v < 8; ++v) {
                float r = acc[nb][v] + bv;
                if (ACT == 1) r = fmaxf(r, 0.0f);
                else if (ACT == 2) r = fmaxf(r, 0.0f) + log1pf(expf(-fabsf(r)));
                C[(long)(mrow + v) * ldc + gn] = r;
            }
        }
    }
}

// ---------------------------------------------------------------------------
// Exact top-k (k = n/2) via 4-pass radix select + ordered index emission.
// One 1024-thread block per batch. Emits ascending, tie-stable indices.
// Also writes y = H @ (p/||p||).
// ---------------------------------------------------------------------------
__global__ __launch_bounds__(1024)
void topk_pool_kernel(const float* __restrict__ H, long strideH,
                      const float* __restrict__ p, int c,
                      int n, int k,
                      float* __restrict__ y, long strideY,
                      int* __restrict__ outIdx, long strideIdx)
{
    const int b = blockIdx.x;
    H      += (long)b * strideH;
    y      += (long)b * strideY;
    outIdx += (long)b * strideIdx;
    const int tid   = threadIdx.x;
    const int items = n >> 10;              // 4 (n=4096) or 2 (n=2048)

    __shared__ float sp[128];
    __shared__ float s_inv;
    __shared__ int   hist[256];
    __shared__ int   s_sel, s_krem;
    __shared__ int   sGT[1024], sEQ[1024];

    if (tid < 128) sp[tid] = (tid < c) ? p[tid] : 0.0f;
    __syncthreads();
    if (tid == 0) {
        float s = 0.0f;
        for (int j = 0; j < c; ++j) s += sp[j] * sp[j];
        s_inv = 1.0f / sqrtf(s);
    }
    __syncthreads();
    const float inv = s_inv;

    unsigned key[4];
    const int base = tid * items;
    for (int it = 0; it < items; ++it) {
        const float* row = H + (long)(base + it) * c;
        float d = 0.0f;
        for (int j = 0; j < c; ++j) d += row[j] * sp[j];
        d *= inv;
        y[base + it] = d;
        const unsigned u = __float_as_uint(d);
        key[it] = u ^ ((u & 0x80000000u) ? 0xFFFFFFFFu : 0x80000000u); // order-preserving
    }

    // radix select: find k-th largest key value T
    unsigned prefix = 0;
    int krem = k;
    for (int shift = 24; shift >= 0; shift -= 8) {
        const unsigned maskHigh = (shift == 24) ? 0u : (0xFFFFFFFFu << (shift + 8));
        if (tid < 256) hist[tid] = 0;
        __syncthreads();
        for (int it = 0; it < items; ++it)
            if ((key[it] & maskHigh) == (prefix & maskHigh))
                atomicAdd(&hist[(key[it] >> shift) & 255], 1);
        __syncthreads();
        if (tid == 0) {
            int cum = 0, bsel = 0;
            for (int bb = 255; bb >= 0; --bb) {
                cum += hist[bb];
                if (cum >= krem) { bsel = bb; break; }
            }
            s_sel  = bsel;
            s_krem = krem - (cum - hist[bsel]);
        }
        __syncthreads();
        prefix |= ((unsigned)s_sel) << shift;
        krem = s_krem;
        __syncthreads();
    }
    const unsigned T   = prefix;
    const int eqNeeded = krem;

    // ordered emission: all keys > T, plus first eqNeeded keys == T (index order)
    int cGT = 0, cEQ = 0;
    for (int it = 0; it < items; ++it) {
        if (key[it] > T) ++cGT;
        else if (key[it] == T) ++cEQ;
    }
    sGT[tid] = cGT; sEQ[tid] = cEQ;
    __syncthreads();
    for (int off = 1; off < 1024; off <<= 1) {
        int a = 0, e = 0;
        if (tid >= off) { a = sGT[tid - off]; e = sEQ[tid - off]; }
        __syncthreads();
        if (tid >= off) { sGT[tid] += a; sEQ[tid] += e; }
        __syncthreads();
    }
    int g = sGT[tid] - cGT;
    int e = sEQ[tid] - cEQ;
    for (int it = 0; it < items; ++it) {
        const int i = base + it;
        if (key[it] > T) {
            const int epos = (e < eqNeeded) ? e : eqNeeded;
            outIdx[g + epos] = i;
            ++g;
        } else if (key[it] == T) {
            if (e < eqNeeded) outIdx[g + e] = i;
            ++e;
        }
    }
}

// ---------------------------------------------------------------------------
// Hp[i,:] = H[idx[i],:] * tanh(y[idx[i]])
// ---------------------------------------------------------------------------
__global__ void gather_gate_kernel(const float* __restrict__ H, long sH,
                                   const float* __restrict__ y, long sY,
                                   const int* __restrict__ idx, long sIdx,
                                   float* __restrict__ Hp, long sHp,
                                   int k, int c)
{
    const int b = blockIdx.y;
    const int t = blockIdx.x * 256 + threadIdx.x;
    if (t >= k * c) return;
    const int i = t / c, j = t % c;
    const int src = idx[(long)b * sIdx + i];
    Hp[(long)b * sHp + (long)i * c + j] =
        H[(long)b * sH + (long)src * c + j] * tanhf(y[(long)b * sY + src]);
}

// gidx[i] = idx1[idx2[i]]
__global__ void compose_idx_kernel(const int* __restrict__ idx1,
                                   const int* __restrict__ idx2,
                                   int* __restrict__ gidx, int k)
{
    const int b = blockIdx.y;
    const int t = blockIdx.x * 256 + threadIdx.x;
    if (t < k) gidx[(long)b * k + t] = idx1[(long)b * 2048 + idx2[(long)b * k + t]];
}

// z = mean + exp(0.5*logvar)*eps ; also emit mean/logvar outputs
__global__ void reparam_kernel(const float* __restrict__ HL, long sHL,
                               const float* __restrict__ eps,
                               float* __restrict__ z,
                               float* __restrict__ mean_o, float* __restrict__ logvar_o)
{
    const int b = blockIdx.y;
    const int t = blockIdx.x * 256 + threadIdx.x;   // < 1024*32
    const int i = t >> 5, j = t & 31;
    const float m  = HL[(long)b * sHL + (long)i * 64 + j];
    const float lv = HL[(long)b * sHL + (long)i * 64 + 32 + j];
    const long  o  = (long)b * 32768 + t;
    z[o]        = m + expf(0.5f * lv) * eps[o];
    mean_o[o]   = m;
    logvar_o[o] = lv;
}

// ---------------------------------------------------------------------------
// Host-side launcher
// ---------------------------------------------------------------------------
template <int ACT>
static inline void launch_gemm_act(hipStream_t s,
                                   const float* A, long sA, int lda,
                                   const int* ridx, const int* cidx,
                                   const float* Bm, long sB, int ldb,
                                   const float* bias,
                                   float* C, long sC, int ldc,
                                   int M, int K, int Nc)
{
    dim3 g((M + 127) / 128, (Nc + 63) / 64, 8), blk(256);
    const bool rg = (ridx != nullptr), cg = (cidx != nullptr);
    if (rg && cg)
        gemm_bf16_wmma<ACT, true,  true ><<<g, blk, 0, s>>>(A, sA, lda, ridx, cidx, Bm, sB, ldb, bias, C, sC, ldc, M, K, Nc);
    else if (rg)
        gemm_bf16_wmma<ACT, true,  false><<<g, blk, 0, s>>>(A, sA, lda, ridx, cidx, Bm, sB, ldb, bias, C, sC, ldc, M, K, Nc);
    else if (cg)
        gemm_bf16_wmma<ACT, false, true ><<<g, blk, 0, s>>>(A, sA, lda, ridx, cidx, Bm, sB, ldb, bias, C, sC, ldc, M, K, Nc);
    else
        gemm_bf16_wmma<ACT, false, false><<<g, blk, 0, s>>>(A, sA, lda, ridx, cidx, Bm, sB, ldb, bias, C, sC, ldc, M, K, Nc);
}

static inline void launch_gemm(hipStream_t s, int act,
                               const float* A, long sA, int lda,
                               const int* ridx, const int* cidx,
                               const float* Bm, long sB, int ldb,
                               const float* bias,
                               float* C, long sC, int ldc,
                               int M, int K, int Nc)
{
    if (act == 0)      launch_gemm_act<0>(s, A, sA, lda, ridx, cidx, Bm, sB, ldb, bias, C, sC, ldc, M, K, Nc);
    else if (act == 1) launch_gemm_act<1>(s, A, sA, lda, ridx, cidx, Bm, sB, ldb, bias, C, sC, ldc, M, K, Nc);
    else               launch_gemm_act<2>(s, A, sA, lda, ridx, cidx, Bm, sB, ldb, bias, C, sC, ldc, M, K, Nc);
}

extern "C" void kernel_launch(void* const* d_in, const int* in_sizes, int n_in,
                              void* d_out, int out_size, void* d_ws, size_t ws_size,
                              hipStream_t stream)
{
    (void)in_sizes; (void)n_in; (void)out_size; (void)ws_size;
    const int Bz = 8;

    const float* x    = (const float*)d_in[0];
    const float* eps  = (const float*)d_in[1];
    const float* A0   = (const float*)d_in[2];
    const float* W1   = (const float*)d_in[3];
    const float* b1   = (const float*)d_in[4];
    const float* p1   = (const float*)d_in[5];
    const float* W2   = (const float*)d_in[6];
    const float* b2   = (const float*)d_in[7];
    const float* p2   = (const float*)d_in[8];
    const float* Wlat = (const float*)d_in[9];
    const float* blat = (const float*)d_in[10];
    const float* Wd0  = (const float*)d_in[11];
    const float* bd0  = (const float*)d_in[12];
    const float* Wd1  = (const float*)d_in[13];
    const float* bd1  = (const float*)d_in[14];
    const float* Wd2  = (const float*)d_in[15];
    const float* bd2  = (const float*)d_in[16];
    const float* Wout = (const float*)d_in[17];
    const float* bout = (const float*)d_in[18];

    float* out       = (float*)d_out;                        // [8,4096,64]
    float* outMean   = out + (size_t)Bz * 4096 * 64;         // [8,1024,32]
    float* outLogvar = outMean + (size_t)Bz * 1024 * 32;     // [8,1024,32]

    // workspace layout (floats)
    float* ws    = (float*)d_ws;
    float* big0  = ws;                                  // Bz*524288
    float* big1  = big0 + (size_t)Bz * 524288;          // Bz*524288
    float* gate  = big1 + (size_t)Bz * 524288;          // Bz*262144
    float* y1    = gate + (size_t)Bz * 262144;          // Bz*4096
    float* y2    = y1 + (size_t)Bz * 4096;              // Bz*2048
    float* zbuf  = y2 + (size_t)Bz * 2048;              // Bz*32768
    int*   idx1  = (int*)(zbuf + (size_t)Bz * 32768);   // Bz*2048
    int*   idx2  = idx1 + (size_t)Bz * 2048;            // Bz*1024
    int*   gidx2 = idx2 + (size_t)Bz * 1024;            // Bz*1024

    // ---- encoder level 0: H1 = relu(A0 @ (x @ W1) + b1) -------------------
    launch_gemm(stream, 0, x, 4096L * 64, 64, nullptr, nullptr, W1, 0, 128, nullptr,
                big0, 524288, 128, 4096, 64, 128);
    launch_gemm(stream, 1, A0, 0, 4096, nullptr, nullptr, big0, 524288, 128, b1,
                big1, 524288, 128, 4096, 4096, 128);

    // ---- pool 1 -----------------------------------------------------------
    topk_pool_kernel<<<Bz, 1024, 0, stream>>>(big1, 524288, p1, 128, 4096, 2048,
                                              y1, 4096, idx1, 2048);
    gather_gate_kernel<<<dim3(1024, Bz), 256, 0, stream>>>(big1, 524288, y1, 4096,
                                                           idx1, 2048, gate, 262144, 2048, 128);

    // ---- encoder level 1: H2 = relu(A1 @ (Hp1 @ W2) + b2), A1 = A0[idx1][:,idx1]
    launch_gemm(stream, 0, gate, 262144, 128, nullptr, nullptr, W2, 0, 64, nullptr,
                big0, 524288, 64, 2048, 128, 64);
    launch_gemm(stream, 1, A0, 0, 4096, idx1, idx1, big0, 524288, 64, b2,
                big1, 524288, 64, 2048, 2048, 64);

    // ---- pool 2 -----------------------------------------------------------
    topk_pool_kernel<<<Bz, 1024, 0, stream>>>(big1, 524288, p2, 64, 2048, 1024,
                                              y2, 2048, idx2, 1024);
    compose_idx_kernel<<<dim3(4, Bz), 256, 0, stream>>>(idx1, idx2, gidx2, 1024);
    gather_gate_kernel<<<dim3(256, Bz), 256, 0, stream>>>(big1, 524288, y2, 2048,
                                                          idx2, 1024, gate, 262144, 1024, 64);

    // ---- latent: HL = A2 @ (Hp2 @ Wlat) + blat, A2 = A0[gidx2][:,gidx2] ---
    launch_gemm(stream, 0, gate, 262144, 64, nullptr, nullptr, Wlat, 0, 64, nullptr,
                big0, 524288, 64, 1024, 64, 64);
    launch_gemm(stream, 0, A0, 0, 4096, gidx2, gidx2, big0, 524288, 64, blat,
                big1, 524288, 64, 1024, 1024, 64);
    reparam_kernel<<<dim3(128, Bz), 256, 0, stream>>>(big1, 524288, eps, zbuf,
                                                      outMean, outLogvar);

    // ---- decoder level 2: D0 = relu(A2 @ (z @ Wd0) + bd0) -----------------
    launch_gemm(stream, 0, zbuf, 32768, 32, nullptr, nullptr, Wd0, 0, 64, nullptr,
                big0, 524288, 64, 1024, 32, 64);
    launch_gemm(stream, 1, A0, 0, 4096, gidx2, gidx2, big0, 524288, 64, bd0,
                big1, 524288, 64, 1024, 1024, 64);

    // ---- decoder level 1: D1 = relu(A0[idx1][:,gidx2] @ (D0 @ Wd1) + bd1) --
    // (unpool folded into the column gather: K = 1024 instead of 2048)
    launch_gemm(stream, 0, big1, 524288, 64, nullptr, nullptr, Wd1, 0, 96, nullptr,
                big0, 524288, 96, 1024, 64, 96);
    launch_gemm(stream, 1, A0, 0, 4096, idx1, gidx2, big0, 524288, 96, bd1,
                big1, 524288, 96, 2048, 1024, 96);

    // ---- decoder level 0: D2 = relu(A0[:, idx1] @ (D1 @ Wd2) + bd2) --------
    launch_gemm(stream, 0, big1, 524288, 96, nullptr, nullptr, Wd2, 0, 128, nullptr,
                big0, 524288, 128, 2048, 96, 128);
    launch_gemm(stream, 1, A0, 0, 4096, nullptr, idx1, big0, 524288, 128, bd2,
                big1, 524288, 128, 4096, 2048, 128);

    // ---- output: OUT = softplus(A0 @ (D2 @ Wout) + bout) -------------------
    launch_gemm(stream, 0, big1, 524288, 128, nullptr, nullptr, Wout, 0, 64, nullptr,
                gate, 262144, 64, 4096, 128, 64);
    launch_gemm(stream, 2, A0, 0, 4096, nullptr, nullptr, gate, 262144, 64, bout,
                out, 262144, 64, 4096, 4096, 64);
}
